// RNN_2439541424713
// MI455X (gfx1250) — compile-verified, matrix operand order checked by reading, not run on hardware
//
#include <hip/hip_runtime.h>
#include <hip/hip_bf16.h>
#include <math.h>

// ---------------------------------------------------------------------------
// CDNA5 (gfx1250) RNN forward: bf16 WMMA GEMMs + sequential tanh recurrence.
// Recurrence step stages w_hh tiles into LDS via the Tensor Data Mover (TDM).
// ---------------------------------------------------------------------------

typedef __attribute__((ext_vector_type(16))) __bf16 v16bf;
typedef __attribute__((ext_vector_type(8)))  __bf16 v8bf;
typedef __attribute__((ext_vector_type(8)))  float  v8f;
typedef __attribute__((ext_vector_type(4)))  unsigned int v4u;
typedef __attribute__((ext_vector_type(8)))  int          v8i;
typedef __attribute__((ext_vector_type(4)))  int          v4i;

#define BDIM 256   // 8 waves (wave32)

// LDS row layout for staged weights: 2048 data bytes + 16B pad after each
// 1024B  ->  row stride 2080 B (= 520 dwords == 8 mod 64: every bank hit
// exactly 2x for a full-wave b128 read -- the minimum possible).
#define LDS_ROW_BYTES   2080
#define LDS_STAGE_ROWS  32

// Load one 16x32 bf16 fragment (A-layout, also B-layout when weights are
// pre-transposed): lane holds row lane%16, k-base (lane/16)*8, VGPRs 0-3 =
// k..k+7, VGPRs 4-7 = k+16..k+23.  Two 16-B loads.
__device__ __forceinline__ v16bf load_frag16x32(const __bf16* p) {
    v8bf lo = *reinterpret_cast<const v8bf*>(p);
    v8bf hi = *reinterpret_cast<const v8bf*>(p + 16);
    return __builtin_shufflevector(lo, hi,
        0,1,2,3,4,5,6,7,8,9,10,11,12,13,14,15);
}

__device__ __forceinline__ v16bf combine_frag(v8bf lo, v8bf hi) {
    return __builtin_shufflevector(lo, hi,
        0,1,2,3,4,5,6,7,8,9,10,11,12,13,14,15);
}

// Byte offset inside a padded LDS row (pad 16B inserted after each 1024B).
__device__ __forceinline__ int pad_off(int b) {
    return b + ((b >= 1024) ? 16 : 0);
}

// ---------------------------------------------------------------------------
// Big parallel GEMM:  C[M][N] = A[M][K](bf16) * BT[N][K]^T(bf16) + bias[N]
// Block covers 64 (M) x 128 (N); 8 waves in 4x2; each wave: 16x64 (4 accs).
// ---------------------------------------------------------------------------
template<int NACC>
__global__ __launch_bounds__(BDIM) void gemm_bias_wmma(
    const __bf16* __restrict__ A,
    const __bf16* __restrict__ BT,
    const float*  __restrict__ bias,
    float*        __restrict__ C,
    int M, int N, int K)
{
    const int wave = threadIdx.x >> 5;
    const int lane = threadIdx.x & 31;
    const int l16  = lane & 15;
    const int lh   = lane >> 4;

    const int m0 = blockIdx.y * 64 + (wave & 3) * 16;
    const int n0 = (blockIdx.x * 2 + (wave >> 2)) * (16 * NACC);

    v8f acc[NACC] = {};

    const __bf16* ap = A  + (size_t)(m0 + l16) * K + lh * 8;
    const __bf16* bp = BT + (size_t)(n0 + l16) * K + lh * 8;

    for (int k = 0; k < K; k += 32) {
        v16bf a = load_frag16x32(ap + k);
#pragma unroll
        for (int j = 0; j < NACC; ++j) {
            v16bf b = load_frag16x32(bp + (size_t)(j * 16) * K + k);
            acc[j] = __builtin_amdgcn_wmma_f32_16x16x32_bf16(
                false, a, false, b, (short)0, acc[j], false, false);
        }
    }

#pragma unroll
    for (int j = 0; j < NACC; ++j) {
        const int   n  = n0 + j * 16 + l16;
        const float bv = bias[n];
#pragma unroll
        for (int v = 0; v < 8; ++v) {
            const int m = m0 + v + 8 * lh;
            C[(size_t)m * N + n] = acc[j][v] + bv;
        }
    }
}

// ---------------------------------------------------------------------------
// One recurrence step:  h_new = tanh(xw[:,t,:] + h_prev @ w_hh)
// Block covers M=64 (all batches) x N=32.  The 32 rows of w_hh^T this block
// needs (64 KB) are staged into LDS once via TDM (tensor_load_to_lds with
// LDS padding for bank-conflict-free b128 reads), then each wave runs the
// K=H WMMA loop with A from global (L2-resident h) and B from LDS.
// ---------------------------------------------------------------------------
__global__ __launch_bounds__(BDIM) void rnn_step_wmma(
    const __bf16* __restrict__ Hprev,   // [64][H] bf16
    const __bf16* __restrict__ WT,      // [H][H] bf16 (w_hh transposed)
    const float*  __restrict__ XW,      // [B*T][H] f32 (bias folded in)
    float*        __restrict__ OUT,     // [B*T][H] f32 hidden_seq
    __bf16*       __restrict__ Hnext,   // [64][H] bf16
    int t, int T, int Hdim)
{
    __shared__ __align__(16) char ldsB[LDS_STAGE_ROWS * LDS_ROW_BYTES];

    const int wave = threadIdx.x >> 5;
    const int lane = threadIdx.x & 31;
    const int l16  = lane & 15;
    const int lh   = lane >> 4;

    const int n0b = blockIdx.x * 32;                      // block n-base
    const int m0  = (wave & 3) * 16;                      // batch tile
    const int nt  = (wave >> 2);                          // n-tile in block
    const int n0  = n0b + nt * 16;

    // ---- TDM: stage WT rows [n0b, n0b+32) x K=Hdim into padded LDS ----
    if (threadIdx.x == 0) {
        const unsigned long long gaddr =
            (unsigned long long)(const void*)(WT + (size_t)n0b * Hdim);
        const unsigned lds_addr = (unsigned)(size_t)(void*)ldsB;

        v4u g0;
        g0[0] = 1u;                                  // count=1, user mode
        g0[1] = lds_addr;                            // lds_addr
        g0[2] = (unsigned)(gaddr & 0xFFFFFFFFull);   // global_addr[31:0]
        g0[3] = (unsigned)((gaddr >> 32) & 0x1FFFFFFull) | (2u << 30); // type=2

        v8i g1;
        g1[0] = (int)((1u << 16)      // data_size = 2 bytes
                    | (1u << 20)      // pad_enable
                    | (7u << 22)      // pad_interval: 256 dwords (1024 B)
                    | (3u << 25));    // pad_amount:   4 dwords (16 B)
        g1[1] = (int)(((unsigned)Hdim & 0xFFFFu) << 16);          // tensor_dim0 lo16
        g1[2] = (int)((((unsigned)Hdim >> 16) & 0xFFFFu)
                    | (((unsigned)Hdim & 0xFFFFu) << 16));        // dim0 hi / dim1 lo
        g1[3] = (int)(((((unsigned)Hdim) >> 16) & 0xFFFFu)
                    | (((unsigned)Hdim & 0xFFFFu) << 16));        // dim1 hi / tile_dim0
        g1[4] = (int)LDS_STAGE_ROWS;                              // tile_dim1=32, tile_dim2=0
        g1[5] = (int)Hdim;                                        // tensor_dim0_stride lo32
        g1[6] = 0;                                                // stride0 hi / stride1 lo
        g1[7] = 0;

        v4i gz4 = {0, 0, 0, 0};                      // 2D: groups 2/3 unused
        v8i gz8 = {0, 0, 0, 0, 0, 0, 0, 0};
        __builtin_amdgcn_tensor_load_to_lds(g0, g1, gz4, gz4, gz8, 0);
        __builtin_amdgcn_s_wait_tensorcnt(0);
    }
    __syncthreads();

    v8f acc = {};
    const __bf16* ap    = Hprev + (size_t)(m0 + l16) * Hdim + lh * 8;
    const char*   lbase = ldsB + (size_t)(nt * 16 + l16) * LDS_ROW_BYTES;

    for (int k = 0; k < Hdim; k += 32) {
        const int b0 = 2 * k + lh * 16;        // first 16-B chunk (data bytes)
        v8bf blo = *reinterpret_cast<const v8bf*>(lbase + pad_off(b0));
        v8bf bhi = *reinterpret_cast<const v8bf*>(lbase + pad_off(b0 + 32));
        v16bf a  = load_frag16x32(ap + k);
        v16bf b  = combine_frag(blo, bhi);
        acc = __builtin_amdgcn_wmma_f32_16x16x32_bf16(
            false, a, false, b, (short)0, acc, false, false);
    }

    const int n = n0 + l16;
#pragma unroll
    for (int v = 0; v < 8; ++v) {
        const int    b   = m0 + v + 8 * lh;
        const size_t idx = ((size_t)b * T + t) * Hdim + n;
        const float  val = acc[v] + XW[idx];
        const float  th  = tanhf(val);
        OUT[idx] = th;
        Hnext[(size_t)b * Hdim + n] = (__bf16)th;
    }
}

// ---------------------------------------------------------------------------
// Prep / epilogue helpers
// ---------------------------------------------------------------------------
__global__ void cvt_bf16(const float* __restrict__ in,
                         __bf16* __restrict__ out, size_t n)
{
    size_t tid = (size_t)blockIdx.x * blockDim.x + threadIdx.x;
    if (tid < n) out[tid] = (__bf16)in[tid];
}

// out[n*K + k] = (bf16) in[k*N + n]   (in is [K][N] row-major f32)
__global__ void transpose_cvt_bf16(const float* __restrict__ in,
                                   __bf16* __restrict__ out, int K, int N)
{
    size_t tid = (size_t)blockIdx.x * blockDim.x + threadIdx.x;
    size_t total = (size_t)K * N;
    if (tid >= total) return;
    int n = (int)(tid / K);
    int k = (int)(tid % K);
    out[tid] = (__bf16)in[(size_t)k * N + n];
}

__global__ void bias_sum(const float* __restrict__ a,
                         const float* __restrict__ b,
                         float* __restrict__ out, int n)
{
    int tid = blockIdx.x * blockDim.x + threadIdx.x;
    if (tid < n) out[tid] = a[tid] + b[tid];
}

__global__ void zero_bf16(__bf16* __restrict__ p, int n)
{
    int tid = blockIdx.x * blockDim.x + threadIdx.x;
    if (tid < n) p[tid] = (__bf16)0.0f;
}

__global__ void copy_last(const float* __restrict__ seq,
                          float* __restrict__ h_t, int B, int T, int Hdim)
{
    int tid = blockIdx.x * blockDim.x + threadIdx.x;
    if (tid >= B * Hdim) return;
    int b = tid / Hdim;
    int n = tid % Hdim;
    h_t[tid] = seq[((size_t)b * T + (T - 1)) * Hdim + n];
}

// ---------------------------------------------------------------------------
// Host entry
// ---------------------------------------------------------------------------
extern "C" void kernel_launch(void* const* d_in, const int* in_sizes, int n_in,
                              void* d_out, int out_size, void* d_ws, size_t ws_size,
                              hipStream_t stream)
{
    (void)in_sizes; (void)n_in; (void)out_size; (void)ws_size;

    const float* x    = (const float*)d_in[0];   // [B,T,I]
    const float* w_ih = (const float*)d_in[1];   // [I,H]
    const float* w_hh = (const float*)d_in[2];   // [H,H]
    const float* b_ih = (const float*)d_in[3];   // [H]
    const float* b_hh = (const float*)d_in[4];   // [H]

    const int B = 64, T = 512, I = 512, H = 1024;
    const size_t M = (size_t)B * T;              // 32768 rows

    // Workspace carve-out (256-B aligned slices).
    char*  ws  = (char*)d_ws;
    size_t off = 0;
    auto carve = [&](size_t bytes) -> char* {
        char* p = ws + off;
        off = (off + bytes + 255) & ~(size_t)255;
        return p;
    };
    float*  xw   = (float*)  carve(M * H * sizeof(float));        // 134 MB
    __bf16* xb   = (__bf16*) carve(M * I * sizeof(__bf16));       //  32 MB
    __bf16* wihT = (__bf16*) carve((size_t)H * I * sizeof(__bf16));
    __bf16* whhT = (__bf16*) carve((size_t)H * H * sizeof(__bf16));
    float*  bias = (float*)  carve(H * sizeof(float));
    __bf16* h0   = (__bf16*) carve((size_t)B * H * sizeof(__bf16));
    __bf16* h1   = (__bf16*) carve((size_t)B * H * sizeof(__bf16));

    float* out = (float*)d_out;                  // hidden_seq [B,T,H]
    float* h_t = out + M * H;                    // h_t [1,B,H]

    // ---- Prep: bf16 conversions, weight transposes, bias, h0 = 0 ----
    {
        size_t n = M * I;
        cvt_bf16<<<(unsigned)((n + 255) / 256), 256, 0, stream>>>(x, xb, n);
    }
    {
        size_t n = (size_t)H * I;
        transpose_cvt_bf16<<<(unsigned)((n + 255) / 256), 256, 0, stream>>>(
            w_ih, wihT, I, H);
    }
    {
        size_t n = (size_t)H * H;
        transpose_cvt_bf16<<<(unsigned)((n + 255) / 256), 256, 0, stream>>>(
            w_hh, whhT, H, H);
    }
    bias_sum<<<(H + 255) / 256, 256, 0, stream>>>(b_ih, b_hh, bias, H);
    zero_bf16<<<(B * H + 255) / 256, 256, 0, stream>>>(h0, B * H);

    // ---- Phase 1: xw = x @ w_ih + (b_ih + b_hh)  (bf16 WMMA, f32 acc) ----
    {
        dim3 grid(H / 128, (unsigned)(M / 64));  // (8, 512)
        gemm_bias_wmma<4><<<grid, BDIM, 0, stream>>>(
            xb, wihT, bias, xw, (int)M, H, I);
    }

    // ---- Phase 2: sequential recurrence, h ping-pong in bf16 ----
    __bf16* hb[2] = { h0, h1 };
    for (int t = 0; t < T; ++t) {
        rnn_step_wmma<<<H / 32, BDIM, 0, stream>>>(
            hb[t & 1], whhT, xw, out, hb[(t + 1) & 1], t, T, H);
    }

    // ---- h_t = hidden_seq[:, T-1, :] ----
    copy_last<<<(B * H + 255) / 256, 256, 0, stream>>>(out, h_t, B, T, H);
}
